// AttentionLSTMDecoder_71425306132807
// MI455X (gfx1250) — matrix operand order repaired and checked by reading, not compiled, not run on hardware
//
#include <hip/hip_runtime.h>
#include <hip/hip_bf16.h>

// ---------------------------------------------------------------------------
// Problem constants (reference: B=32, T_MEM=256, T_TGT=128, E=H=512, V=10000)
// ---------------------------------------------------------------------------
#define BATCH 32
#define TMEM  256
#define TTGT  128
#define EDIM  512
#define HDIM  512
#define VOCAB 10000
#define KCAT  1536          // E (context) + E (x_t) + H (h)
#define GATES 2048          // 4*H, torch order i,f,g,o
#define NTILE_V 625         // 10000 / 16
#define KWAVES 4            // K-split waves in the gates kernel
#define KSLICE (KCAT / KWAVES)   // 384 = 12 K-steps of 32

typedef __attribute__((ext_vector_type(16))) __bf16  v16bf;
typedef __attribute__((ext_vector_type(8)))  float   v8f;
typedef __attribute__((ext_vector_type(4)))  unsigned int u32x4;

union BF16x16 { v16bf v; u32x4 q[2]; };

__device__ __forceinline__ __bf16 f2bf(float f) {
  union { float f; unsigned u; } x; x.f = f;
  unsigned r = x.u + 0x7FFFu + ((x.u >> 16) & 1u);   // round-to-nearest-even
  unsigned short h = (unsigned short)(r >> 16);
  __bf16 b; __builtin_memcpy(&b, &h, 2);
  return b;
}

__device__ __forceinline__ float sigmoidf(float x) {
  return 1.0f / (1.0f + __expf(-x));
}

// ---------------------------------------------------------------------------
// Prep: fuse W_ih (2048x1024) and W_hh (2048x512) into bf16 W_cat (2048x1536)
// ---------------------------------------------------------------------------
__global__ __launch_bounds__(256)
void prep_wcat_kernel(const float* __restrict__ Wih, const float* __restrict__ Whh,
                      __bf16* __restrict__ Wcat) {
  int idx = blockIdx.x * 256 + threadIdx.x;          // < 2048*1536
  int r = idx / KCAT, c = idx % KCAT;
  float v = (c < 1024) ? Wih[(size_t)r * 1024 + c]
                       : Whh[(size_t)r * 512 + (c - 1024)];
  Wcat[idx] = f2bf(v);
}

__global__ __launch_bounds__(256)
void prep_wout_kernel(const float* __restrict__ Wout, __bf16* __restrict__ WoutB) {
  int idx = blockIdx.x * 256 + threadIdx.x;          // < 10000*512
  WoutB[idx] = f2bf(Wout[idx]);
}

// bias = b_ih + b_hh ; h = c = 0
__global__ __launch_bounds__(256)
void prep_state_kernel(const float* __restrict__ bih, const float* __restrict__ bhh,
                       float* __restrict__ bias, float* __restrict__ h,
                       float* __restrict__ c) {
  int idx = blockIdx.x * 256 + threadIdx.x;          // 16384 threads
  if (idx < GATES) bias[idx] = bih[idx] + bhh[idx];
  h[idx] = 0.0f;
  c[idx] = 0.0f;
}

// ---------------------------------------------------------------------------
// Step kernel 1: attention + build bf16 A row = [context | x_t | h]
// one workgroup per batch row, 256 threads
// ---------------------------------------------------------------------------
__global__ __launch_bounds__(256)
void attn_kernel(const float* __restrict__ memory, const float* __restrict__ hbuf,
                 const float* __restrict__ emb_in, const int* __restrict__ tgt,
                 __bf16* __restrict__ Abuf, int t) {
  int b = blockIdx.x;
  int tid = threadIdx.x;

  __shared__ float hs[EDIM];      // h row (fp32)
  __shared__ float sraw[TMEM];    // raw scores
  __shared__ float sc[TMEM];      // reduction scratch / attn weights

  hs[tid]       = hbuf[b * HDIM + tid];
  hs[tid + 256] = hbuf[b * HDIM + 256 + tid];
  __syncthreads();

  // score for key tm = tid : dot(h, memory[b][tm]) * 1/sqrt(E)
  const float* mrow = memory + ((size_t)b * TMEM + tid) * EDIM;
  float s = 0.0f;
  #pragma unroll 8
  for (int e = 0; e < EDIM; ++e) s = fmaf(hs[e], mrow[e], s);
  s *= 0.044194173824159216f;     // 1/sqrt(512)

  // stable softmax over 256
  sraw[tid] = s; sc[tid] = s; __syncthreads();
  for (int off = 128; off > 0; off >>= 1) {
    if (tid < off) sc[tid] = fmaxf(sc[tid], sc[tid + off]);
    __syncthreads();
  }
  float mx = sc[0]; __syncthreads();
  float p = __expf(sraw[tid] - mx);
  sc[tid] = p; __syncthreads();
  for (int off = 128; off > 0; off >>= 1) {
    if (tid < off) sc[tid] += sc[tid + off];
    __syncthreads();
  }
  float rsum = 1.0f / sc[0]; __syncthreads();
  sc[tid] = p * rsum;             // attention weight
  __syncthreads();

  // context[e] = sum_tm attn[tm] * memory[b][tm][e]  (coalesced over e)
  const float* mb = memory + (size_t)b * TMEM * EDIM;
  float c0 = 0.0f, c1 = 0.0f;
  #pragma unroll 4
  for (int tm = 0; tm < TMEM; ++tm) {
    float a = sc[tm];
    c0 = fmaf(a, mb[(size_t)tm * EDIM + tid],       c0);
    c1 = fmaf(a, mb[(size_t)tm * EDIM + 256 + tid], c1);
  }

  __bf16* Arow = Abuf + (size_t)b * KCAT;
  Arow[tid]       = f2bf(c0);
  Arow[256 + tid] = f2bf(c1);

  // x_t = emb_in[tgt[b][t]]
  int tok = tgt[b * TTGT + t];
  const float* er = emb_in + (size_t)tok * EDIM;
  Arow[512 + tid] = f2bf(er[tid]);
  Arow[768 + tid] = f2bf(er[256 + tid]);

  // h (bf16 copy for the gates GEMM)
  Arow[1024 + tid] = f2bf(hs[tid]);
  Arow[1280 + tid] = f2bf(hs[256 + tid]);
}

// ---------------------------------------------------------------------------
// Step kernel 2: gates = A(32x1536) @ W_cat^T (+bias), fused LSTM pointwise.
// 32 workgroups x 4 waves. Each wave owns a 384-wide K-slice (12 K-steps,
// 96 WMMAs); partial accumulators are reduced through LDS; waves 0/1 apply
// the LSTM nonlinearity for M-tile 0/1 entirely from registers.
// ---------------------------------------------------------------------------
__global__ __launch_bounds__(32 * KWAVES)
void gates_lstm_kernel(const __bf16* __restrict__ Wcat, const float* __restrict__ bias,
                       const __bf16* __restrict__ Abuf,
                       float* __restrict__ hbuf, float* __restrict__ cbuf,
                       __bf16* __restrict__ hbf) {
  __shared__ float part[KWAVES * 2 * 4 * 256];   // [wave][mt][gate][lane*8+r] 32 KB

  int j0   = blockIdx.x * 16;        // hidden column block
  int tid  = threadIdx.x;
  int wave = tid >> 5;
  int lane = tid & 31;
  int hi   = lane >> 4;              // 0 | 1 (lane half)
  int mrow = lane & 15;              // A row within tile
  int khA  = hi * 8;                 // A: lanes 0-15 -> K{0-7,16-23}; 16-31 -> K{8-15,24-31}
  int ncol = lane & 15;              // B column
  int kbB  = hi * 8;                 // B: lanes 0-15 -> K{0-7,16-23}; 16-31 -> K{8-15,24-31}

  v8f acc[2][4];
  #pragma unroll
  for (int mt = 0; mt < 2; ++mt)
    #pragma unroll
    for (int g = 0; g < 4; ++g) acc[mt][g] = (v8f){0,0,0,0,0,0,0,0};

  int kbeg = wave * KSLICE;
  for (int kt = 0; kt < KSLICE / 32; ++kt) {     // 12 iterations
    int k = kbeg + kt * 32;
    BF16x16 a[2];
    #pragma unroll
    for (int mt = 0; mt < 2; ++mt) {
      const __bf16* ap = Abuf + (size_t)(mt * 16 + mrow) * KCAT + k + khA;
      a[mt].q[0] = *(const u32x4*)ap;            // K khA+0..7
      a[mt].q[1] = *(const u32x4*)(ap + 16);     // K khA+16..23
    }
    #pragma unroll
    for (int g = 0; g < 4; ++g) {
      const __bf16* bp = Wcat + (size_t)(g * HDIM + j0 + ncol) * KCAT + k + kbB;
      if (kt + 1 < KSLICE / 32)
        __builtin_prefetch(bp + 32, 0, 1);       // global_prefetch next K-slice
      BF16x16 bm;
      bm.q[0] = *(const u32x4*)bp;               // K kbB+0..7
      bm.q[1] = *(const u32x4*)(bp + 16);        // K kbB+16..23
      #pragma unroll
      for (int mt = 0; mt < 2; ++mt)
        acc[mt][g] = __builtin_amdgcn_wmma_f32_16x16x32_bf16(
            false, a[mt].v, false, bm.v, (short)0, acc[mt][g], false, false);
    }
  }

  // dump partial accumulators to LDS
  #pragma unroll
  for (int mt = 0; mt < 2; ++mt)
    #pragma unroll
    for (int g = 0; g < 4; ++g)
      #pragma unroll
      for (int r = 0; r < 8; ++r)
        part[(((wave * 2 + mt) * 4 + g) << 8) + lane * 8 + r] = acc[mt][g][r];
  __syncthreads();

  // waves 0/1: reduce K-partials for M-tile (wave), apply LSTM pointwise
  if (wave < 2) {
    int mt  = wave;
    int col = j0 + ncol;
    float bi = bias[0 * HDIM + col];
    float bf = bias[1 * HDIM + col];
    float bg = bias[2 * HDIM + col];
    float bo = bias[3 * HDIM + col];

    #pragma unroll
    for (int r = 0; r < 8; ++r) {
      float s0 = 0.f, s1 = 0.f, s2 = 0.f, s3 = 0.f;
      #pragma unroll
      for (int w = 0; w < KWAVES; ++w) {
        int base = ((w * 2 + mt) << 10) + lane * 8 + r;   // (w*2+mt)*4*256
        s0 += part[base];
        s1 += part[base + 256];
        s2 += part[base + 512];
        s3 += part[base + 768];
      }
      float gi = s0 + bi;
      float gf = s1 + bf;
      float gg = s2 + bg;
      float go = s3 + bo;
      int brow = mt * 16 + r + hi * 8;          // batch row (C/D layout)
      size_t idx = (size_t)brow * HDIM + col;
      float cn = sigmoidf(gf) * cbuf[idx] + sigmoidf(gi) * tanhf(gg);
      float hn = sigmoidf(go) * tanhf(cn);
      cbuf[idx] = cn;
      hbuf[idx] = hn;
      hbf[idx]  = f2bf(hn);
    }
  }
}

// ---------------------------------------------------------------------------
// Step kernel 3: logits (32x10000) = h(32x512) @ W_out^T + b_out.
// One wave per 16 vocab columns; 625 workgroups, 32 WMMAs each.
// ---------------------------------------------------------------------------
__global__ __launch_bounds__(32)
void logits_kernel(const __bf16* __restrict__ hbf, const __bf16* __restrict__ WoutB,
                   const float* __restrict__ bout, float* __restrict__ out, int t) {
  int n0   = blockIdx.x * 16;
  int lane = threadIdx.x;
  int hi   = lane >> 4;
  int mrow = lane & 15;
  int khA  = hi * 8;
  int ncol = lane & 15;
  int kbB  = hi * 8;

  v8f acc0 = (v8f){0,0,0,0,0,0,0,0};
  v8f acc1 = (v8f){0,0,0,0,0,0,0,0};

  for (int kt = 0; kt < HDIM / 32; ++kt) {
    int k = kt * 32;
    BF16x16 a0, a1, bm;
    const __bf16* ap0 = hbf + (size_t)mrow * HDIM + k + khA;
    const __bf16* ap1 = hbf + (size_t)(16 + mrow) * HDIM + k + khA;
    a0.q[0] = *(const u32x4*)ap0;  a0.q[1] = *(const u32x4*)(ap0 + 16);
    a1.q[0] = *(const u32x4*)ap1;  a1.q[1] = *(const u32x4*)(ap1 + 16);
    const __bf16* bp = WoutB + (size_t)(n0 + ncol) * HDIM + k + kbB;
    bm.q[0] = *(const u32x4*)bp;   bm.q[1] = *(const u32x4*)(bp + 16);
    acc0 = __builtin_amdgcn_wmma_f32_16x16x32_bf16(false, a0.v, false, bm.v,
                                                   (short)0, acc0, false, false);
    acc1 = __builtin_amdgcn_wmma_f32_16x16x32_bf16(false, a1.v, false, bm.v,
                                                   (short)0, acc1, false, false);
  }

  int col = n0 + ncol;
  float bb = bout[col];
  #pragma unroll
  for (int r = 0; r < 8; ++r) {
    int b0 = r + hi * 8;                       // batch rows 0..15 / 16..31
    out[((size_t)b0        * TTGT + t) * VOCAB + col] = acc0[r] + bb;
    out[((size_t)(b0 + 16) * TTGT + t) * VOCAB + col] = acc1[r] + bb;
  }
}

// ---------------------------------------------------------------------------
// Host launch
// ---------------------------------------------------------------------------
extern "C" void kernel_launch(void* const* d_in, const int* in_sizes, int n_in,
                              void* d_out, int out_size, void* d_ws, size_t ws_size,
                              hipStream_t stream) {
  const float* memory = (const float*)d_in[0];   // (32,256,512)
  const int*   tgt    = (const int*)  d_in[1];   // (32,128)
  const float* emb_in = (const float*)d_in[2];   // (10000,512)
  const float* W_ih   = (const float*)d_in[3];   // (2048,1024)
  const float* W_hh   = (const float*)d_in[4];   // (2048,512)
  const float* b_ih   = (const float*)d_in[5];   // (2048,)
  const float* b_hh   = (const float*)d_in[6];   // (2048,)
  const float* W_out  = (const float*)d_in[7];   // (10000,512)
  const float* b_out  = (const float*)d_in[8];   // (10000,)
  float*       out    = (float*)d_out;           // (32,128,10000)

  // workspace carve-up (all 16B-aligned)
  char* ws = (char*)d_ws;
  __bf16* Wcat  = (__bf16*)ws;  ws += (size_t)GATES * KCAT * 2;   //  6.29 MB
  __bf16* WoutB = (__bf16*)ws;  ws += (size_t)VOCAB * HDIM * 2;   // 10.24 MB
  float*  bias  = (float*) ws;  ws += (size_t)GATES * 4;
  __bf16* Abuf  = (__bf16*)ws;  ws += (size_t)BATCH * KCAT * 2;
  float*  hbuf  = (float*) ws;  ws += (size_t)BATCH * HDIM * 4;
  float*  cbuf  = (float*) ws;  ws += (size_t)BATCH * HDIM * 4;
  __bf16* hbf   = (__bf16*)ws;  ws += (size_t)BATCH * HDIM * 2;

  // one-time prep (weights -> bf16, fused bias, zero state)
  prep_wcat_kernel <<<(GATES * KCAT) / 256, 256, 0, stream>>>(W_ih, W_hh, Wcat);
  prep_wout_kernel <<<(VOCAB * HDIM) / 256, 256, 0, stream>>>(W_out, WoutB);
  prep_state_kernel<<<(BATCH * HDIM) / 256, 256, 0, stream>>>(b_ih, b_hh, bias, hbuf, cbuf);

  // sequential decode: 3 stream-ordered kernels per step
  for (int t = 0; t < TTGT; ++t) {
    attn_kernel      <<<BATCH, 256,          0, stream>>>(memory, hbuf, emb_in, tgt, Abuf, t);
    gates_lstm_kernel<<<HDIM/16, 32*KWAVES,  0, stream>>>(Wcat, bias, Abuf, hbuf, cbuf, hbf);
    logits_kernel    <<<NTILE_V, 32,         0, stream>>>(hbf, WoutB, b_out, out, t);
  }
}
